// NetworkPINeuFlow_14791867367498
// MI455X (gfx1250) — matrix operand order repaired and checked by reading, not compiled, not run on hardware
//
#include <hip/hip_runtime.h>

typedef _Float16 f16;
typedef __attribute__((ext_vector_type(2)))  _Float16 v2h;
typedef __attribute__((ext_vector_type(16))) _Float16 v16h;
typedef __attribute__((ext_vector_type(8)))  float    v8f;

#define HASH_T    (1u << 19)
#define HASH_MASK (HASH_T - 1u)
#define PTS_BLK   128
#define NTHR      256

// ---------------------------------------------------------------------------
// Fragment-native LDS layouts: each lane's 16 f16 operand values live in one
// contiguous 32-byte chunk -> loaded with 2x ds_load_b128 (no u16 gathers).
// ---------------------------------------------------------------------------

// Map (batch row p in 0..127, input channel kk) -> index in A-fragment layout
// [rowtile(8)][ktile(2)][lane(32)][elem(16)].
// ISA A layout (16x32 f16): lane<16 holds K {0..7,16..23}; lane>=16 {8..15,24..31}.
__device__ __forceinline__ int afrag_idx(int p, int kk) {
  int rt = p >> 4;
  int ml = p & 15;
  int kt = kk >> 5;
  int k5 = kk & 31;
  int la = ml + ((k5 & 8) ? 16 : 0);
  int e  = ((k5 & 16) ? 8 : 0) | (k5 & 7);
  return ((rt * 2 + kt) * 32 + la) * 16 + e;
}

// Load one lane's 32B fragment chunk as 2x b128.
__device__ __forceinline__ v16h load_chunk16(const f16* p) {
  union { uint4 u[2]; v16h h; } t;
  const uint4* q = (const uint4*)p;
  t.u[0] = q[0];
  t.u[1] = q[1];
  return t.h;
}

// Single-instruction relu: v_med3_f32(x, 0, +inf). Raw intrinsic -> no
// canonicalize/double-max legalization.
__device__ __forceinline__ float relu1(float r) {
  return __builtin_amdgcn_fmed3f(r, 0.0f, __builtin_inff());
}

// Pack-convert two floats to f16 pair via v_cvt_pk_rtz_f16_f32.
__device__ __forceinline__ v2h pack2(float a, float b) {
  return __builtin_bit_cast(v2h, __builtin_amdgcn_cvt_pkrtz(a, b));
}

// Stage a row-major f32 weight [kreal][ncreal] into B-fragment layout
// [ktile(KT)][ntile(NT)][lane(32)][elem(16)] as f16, zero-padding K and N.
// ISA B layout (32x16 f16): lane<16 holds col N=lane, K {0..7,16..23}; lane>=16 {8..15,24..31}.
template <int KT, int NT>
__device__ __forceinline__ void stage_weight(f16* dst, const float* __restrict__ src,
                                             int kreal, int ncreal, int t) {
  const int total = KT * NT * 32 * 16;
  for (int i = t; i < total; i += NTHR) {
    int e    = i & 15;
    int la   = (i >> 4) & 31;
    int tile = i >> 9;
    int j    = tile % NT;
    int k    = tile / NT;
    int col  = j * 16 + (la & 15);
    int row  = 32 * k + ((la < 16) ? 0 : 8) + (e & 7) + ((e & 8) ? 16 : 0);
    float v  = (row < kreal && col < ncreal) ? src[row * ncreal + col] : 0.0f;
    dst[i] = (f16)v;
  }
}

// In-place [128][64] layer: X = relu(X @ W). Activations in A-fragment layout,
// weights pre-swizzled into B-fragment layout. All loads issue before stores
// (per-wave LDS in-order) so in-place is safe.
template <int KT>
__device__ __forceinline__ void layer64_relu(f16* act, const f16* wf,
                                             int wave, int lane) {
  v16h a[KT];
#pragma unroll
  for (int k = 0; k < KT; ++k)
    a[k] = load_chunk16(act + (((wave * 2 + k) * 32 + lane) << 4));

  v8f acc[4];
#pragma unroll
  for (int j = 0; j < 4; ++j) {
    v8f c = {};
#pragma unroll
    for (int k = 0; k < KT; ++k) {
      v16h b = load_chunk16(wf + (((k * 4 + j) * 32 + lane) << 4));
      c = __builtin_amdgcn_wmma_f32_16x16x32_f16(false, a[k], false, b,
                                                 (short)0, c, false, false);
    }
    acc[j] = c;
  }

  const int col0  = lane & 15;
  const int mbase = wave * 16 + ((lane < 16) ? 0 : 8);
#pragma unroll
  for (int j = 0; j < 4; ++j) {
    const int kk = 16 * j + col0;
#pragma unroll
    for (int v = 0; v < 8; v += 2) {
      // relu via v_med3, pack-convert pairs via v_cvt_pk_rtz_f16_f32;
      // the two halves go out as ds_store_b16 / ds_store_b16_d16_hi.
      v2h pk = pack2(relu1(acc[j][v]), relu1(acc[j][v + 1]));
      act[afrag_idx(mbase + v,     kk)] = pk[0];
      act[afrag_idx(mbase + v + 1, kk)] = pk[1];
    }
  }
}

// [128][64] @ [64][16] -> per-lane C fragment (single N tile, no store).
template <int KT>
__device__ __forceinline__ v8f layer16(const f16* act, const f16* wf,
                                       int wave, int lane) {
  v8f c = {};
#pragma unroll
  for (int k = 0; k < KT; ++k) {
    v16h a = load_chunk16(act + (((wave * 2 + k) * 32 + lane) << 4));
    v16h b = load_chunk16(wf + ((k * 32 + lane) << 4));
    c = __builtin_amdgcn_wmma_f32_16x16x32_f16(false, a, false, b,
                                               (short)0, c, false, false);
  }
  return c;
}

__device__ __forceinline__ float sel3(int i, float x, float y, float z) {
  return (i == 0) ? x : ((i == 1) ? y : z);
}

// ---------------------------------------------------------------------------
// Fused NGP-NeRF kernel: hash-encode -> sigma MLP -> freq-encode -> color MLP
// ---------------------------------------------------------------------------
__global__ __launch_bounds__(NTHR, 1)
void nerf_fused_kernel(const float* __restrict__ xyzt,   // [n][4]
                       const float* __restrict__ dirs,   // [n][3]
                       const float* __restrict__ table,  // [16][T][2]
                       const float* __restrict__ sw0,    // [32][64]
                       const float* __restrict__ sw1,    // [64][64]
                       const float* __restrict__ sw2,    // [64][16]
                       const float* __restrict__ cw0,    // [42][64]
                       const float* __restrict__ cw1,    // [64][64]
                       const float* __restrict__ cw2,    // [64][3]
                       float* __restrict__ out,          // sigma[n] ++ rgb[n][3]
                       int n) {
  __shared__ alignas(16) f16 s_act[8 * 2 * 32 * 16];  // A-frag layout, in-place
  __shared__ alignas(16) f16 s_h2[PTS_BLK * 16];      // geo feats (row-major)
  __shared__ alignas(16) f16 s_ws0[1 * 4 * 32 * 16];
  __shared__ alignas(16) f16 s_ws1[2 * 4 * 32 * 16];
  __shared__ alignas(16) f16 s_ws2[2 * 1 * 32 * 16];
  __shared__ alignas(16) f16 s_wc0[2 * 4 * 32 * 16];
  __shared__ alignas(16) f16 s_wc1[2 * 4 * 32 * 16];
  __shared__ alignas(16) f16 s_wc2[2 * 1 * 32 * 16];

  const int t     = threadIdx.x;
  const int lane  = t & 31;
  const int wave  = t >> 5;
  const int gbase = blockIdx.x * PTS_BLK;

  // ---- phase 0: stage weights into B-fragment layout (f32 -> f16) ----------
  stage_weight<1, 4>(s_ws0, sw0, 32, 64, t);
  stage_weight<2, 4>(s_ws1, sw1, 64, 64, t);
  stage_weight<2, 1>(s_ws2, sw2, 64, 16, t);
  stage_weight<2, 4>(s_wc0, cw0, 42, 64, t);
  stage_weight<2, 4>(s_wc1, cw1, 64, 64, t);
  stage_weight<2, 1>(s_wc2, cw2, 64, 3, t);
  __syncthreads();

  // ---- phase 1: multiresolution hash encoding ------------------------------
  // 2 threads per point: each handles 8 of 16 levels (wave-local rows).
  // Corner loop fully unrolled: 16 independent global_load_b64 in flight per
  // level iteration (this is the latency-bound phase; gathers are L2-resident).
  {
    const int p    = t >> 1;
    const int half = t & 1;
    int gp = gbase + p;
    if (gp >= n) gp = n - 1;  // clamp loads
    const float x0 = xyzt[gp * 4 + 0], x1 = xyzt[gp * 4 + 1];
    const float x2 = xyzt[gp * 4 + 2], x3 = xyzt[gp * 4 + 3];
    const unsigned primes[4] = {1u, 2654435761u, 805459861u, 3674653429u};
    const float klog = 0.23104906018664842f;  // ln(512/16)/15

    for (int li = 0; li < 8; ++li) {
      const int lev = half * 8 + li;
      const float res = floorf(16.0f * __expf((float)lev * klog));
      float pf[4] = {x0 * res, x1 * res, x2 * res, x3 * res};
      float w[4];
      unsigned c0[4];
#pragma unroll
      for (int d = 0; d < 4; ++d) {
        float fl = floorf(pf[d]);
        w[d]  = pf[d] - fl;
        c0[d] = (unsigned)(int)fl;
      }
      const float* tab = table + (size_t)lev * (size_t)HASH_T * 2u;

      // compute all 16 indices/weights, then issue all 16 gathers
      float2 fv[16];
      float  wts[16];
#pragma unroll
      for (int corner = 0; corner < 16; ++corner) {
        unsigned h = 0u;
        float wt = 1.0f;
#pragma unroll
        for (int d = 0; d < 4; ++d) {
          unsigned bit = (unsigned)(corner >> d) & 1u;
          h ^= (c0[d] + bit) * primes[d];
          wt *= bit ? w[d] : (1.0f - w[d]);
        }
        wts[corner] = wt;
        fv[corner]  = *(const float2*)(tab + (size_t)(h & HASH_MASK) * 2u);
      }
      float f0 = 0.0f, f1 = 0.0f;
#pragma unroll
      for (int corner = 0; corner < 16; ++corner) {
        f0 += fv[corner].x * wts[corner];
        f1 += fv[corner].y * wts[corner];
      }
      s_act[afrag_idx(p, lev * 2 + 0)] = (f16)f0;
      s_act[afrag_idx(p, lev * 2 + 1)] = (f16)f1;
    }
  }
  __syncthreads();

  // ---- phase 2: sigma MLP (32 -> 64 -> 64 -> 16) via WMMA ------------------
  layer64_relu<1>(s_act, s_ws0, wave, lane);  // K tile 0 only
  layer64_relu<2>(s_act, s_ws1, wave, lane);
  {
    v8f h = layer16<2>(s_act, s_ws2, wave, lane);
    const int col = lane & 15;
    const int rb  = wave * 16 + ((lane < 16) ? 0 : 8);
#pragma unroll
    for (int v = 0; v < 8; ++v) s_h2[(rb + v) * 16 + col] = (f16)h[v];
    if (col == 0) {  // sigma straight from the fragment (f32)
#pragma unroll
      for (int v = 0; v < 8; ++v) {
        int gp = gbase + rb + v;
        if (gp < n) out[gp] = h[v];
      }
    }
  }
  __syncthreads();

  // ---- phase 3: color input: dirs(3) | sin(12) | cos(12) | geo(15) | 0-pad -
  {
    const int p    = t >> 1;
    const int half = t & 1;
    int gp = gbase + p;
    if (gp >= n) gp = n - 1;
    const float dx = dirs[gp * 3 + 0];
    const float dy = dirs[gp * 3 + 1];
    const float dz = dirs[gp * 3 + 2];
#pragma unroll
    for (int j = 0; j < 32; ++j) {
      const int col = half * 32 + j;
      float v;
      if (col < 3) {
        v = sel3(col, dx, dy, dz);
      } else if (col < 15) {
        int q = col - 3;
        v = __sinf(sel3(q % 3, dx, dy, dz) * (float)(1 << (q / 3)));
      } else if (col < 27) {
        int q = col - 15;
        v = __cosf(sel3(q % 3, dx, dy, dz) * (float)(1 << (q / 3)));
      } else if (col < 42) {
        v = (float)s_h2[p * 16 + (col - 27 + 1)];
      } else {
        v = 0.0f;
      }
      s_act[afrag_idx(p, col)] = (f16)v;
    }
  }
  __syncthreads();

  // ---- phase 4: color MLP (42(pad64) -> 64 -> 64 -> 3(pad16)) --------------
  layer64_relu<2>(s_act, s_wc0, wave, lane);
  layer64_relu<2>(s_act, s_wc1, wave, lane);
  {
    v8f rgb = layer16<2>(s_act, s_wc2, wave, lane);
    const int col = lane & 15;
    const int rb  = wave * 16 + ((lane < 16) ? 0 : 8);
    if (col < 3) {
#pragma unroll
      for (int v = 0; v < 8; ++v) {
        int gp = gbase + rb + v;
        if (gp < n) {
          // sigmoid with fast reciprocal (3 channels only)
          float e = __expf(-rgb[v]);
          out[n + gp * 3 + col] = __builtin_amdgcn_rcpf(1.0f + e);
        }
      }
    }
  }
}

// ---------------------------------------------------------------------------
extern "C" void kernel_launch(void* const* d_in, const int* in_sizes, int n_in,
                              void* d_out, int out_size, void* d_ws, size_t ws_size,
                              hipStream_t stream) {
  (void)n_in; (void)out_size; (void)d_ws; (void)ws_size;
  const float* xyzt  = (const float*)d_in[0];
  const float* dirs  = (const float*)d_in[1];
  const float* table = (const float*)d_in[2];
  const float* sw0   = (const float*)d_in[3];
  const float* sw1   = (const float*)d_in[4];
  const float* sw2   = (const float*)d_in[5];
  const float* cw0   = (const float*)d_in[6];
  const float* cw1   = (const float*)d_in[7];
  const float* cw2   = (const float*)d_in[8];
  float* out = (float*)d_out;

  const int n = in_sizes[0] / 4;
  const int nblk = (n + PTS_BLK - 1) / PTS_BLK;
  nerf_fused_kernel<<<nblk, NTHR, 0, stream>>>(xyzt, dirs, table, sw0, sw1, sw2,
                                               cw0, cw1, cw2, out, n);
}
